// MatchingLoss_57698590654572
// MI455X (gfx1250) — matrix operand order repaired and checked by reading, not compiled, not run on hardware
//
#include <hip/hip_runtime.h>
#include <math.h>

#define BB  8
#define NT  100
#define NP  300
#define CC  92
#define AA  64
#define NTP 112     // NT padded to 16
#define NPP 304     // NP padded to 16
#define EPSF 1e-7f

typedef __attribute__((ext_vector_type(16))) _Float16 v16h;
typedef __attribute__((ext_vector_type(8)))  float    v8f;

// ---------------------------------------------------------------------------
// Prep kernels
// ---------------------------------------------------------------------------

// attribute (0/1 f32) -> padded f16 Y [B][NTP][AA], zero-padded rows
__global__ void prep_y_kernel(const float* __restrict__ attribute, _Float16* __restrict__ Yh) {
    int idx = blockIdx.x * blockDim.x + threadIdx.x;
    const int total = BB * NTP * AA;
    if (idx >= total) return;
    int a = idx % AA; int rem = idx / AA;
    int t = rem % NTP; int b = rem / NTP;
    float v = (t < NT) ? attribute[((size_t)b * NT + t) * AA + a] : 0.0f;
    Yh[idx] = (_Float16)v;
}

// attribute_preds -> G = f1 - f0 (f16, padded [B][NPP][AA]) and S0 = sum_a f0
__global__ void prep_g_kernel(const float* __restrict__ attr_preds,
                              _Float16* __restrict__ Gh, float* __restrict__ S0) {
    int idx = blockIdx.x * blockDim.x + threadIdx.x;   // over B*NPP
    if (idx >= BB * NPP) return;
    int p = idx % NPP; int b = idx / NPP;
    _Float16* grow = Gh + (size_t)idx * AA;
    float s0 = 0.0f;
    if (p < NP) {
        const float* row = attr_preds + ((size_t)b * NP + p) * AA;
        for (int a = 0; a < AA; ++a) {
            float pv = row[a];
            pv = fminf(fmaxf(pv, EPSF), 1.0f - EPSF);
            float f1 = 0.25f * (1.0f - pv) * (1.0f - pv) * (-logf(pv));        // y=1 term
            float f0 = 0.75f * pv * pv * (-logf(1.0f - pv));                    // y=0 term
            grow[a] = (_Float16)(f1 - f0);
            s0 += f0;
        }
    } else {
        for (int a = 0; a < AA; ++a) grow[a] = (_Float16)0.0f;
    }
    S0[idx] = s0;
}

// one-hot category -> index
__global__ void prep_catidx_kernel(const float* __restrict__ category, int* __restrict__ cat_idx) {
    int idx = blockIdx.x * blockDim.x + threadIdx.x;   // over B*NT
    if (idx >= BB * NT) return;
    const float* row = category + (size_t)idx * CC;
    int k = 0;
    for (int c = 0; c < CC; ++c) { if (row[c] > 0.5f) { k = c; break; } }
    cat_idx[idx] = k;
}

// ---------------------------------------------------------------------------
// WMMA kernel: Dot[b][t][p] = sum_a Y[b][t][a] * G[b][p][a]   (K = 64 = 2 x 32)
// One wave32 per 16x16 output tile; v_wmma_f32_16x16x32_f16.
// ---------------------------------------------------------------------------
__global__ void __launch_bounds__(32) attr_wmma_kernel(const _Float16* __restrict__ Yh,
                                                       const _Float16* __restrict__ Gh,
                                                       float* __restrict__ Dot) {
    const int TP = NPP / 16;               // 19
    const int TT = NTP / 16;               // 7
    int id = blockIdx.x;
    int pp = id % TP; id /= TP;
    int tt = id % TT; int b = id / TT;

    int lane = threadIdx.x;
    int m = lane & 15;                     // A-row / B-col / D-col within tile
    int h = lane >> 4;                     // lane half selects K group

    const _Float16* yrow = Yh + ((size_t)b * NTP + tt * 16 + m) * AA;
    const _Float16* grow = Gh + ((size_t)b * NPP + pp * 16 + m) * AA;

    v8f acc = {};
#pragma unroll
    for (int kk = 0; kk < 2; ++kk) {       // K steps of 32
        const int base = kk * 32;
        v16h af, bf;
        // A 16x32 f16 layout: e<8 -> K = base + h*8 + e ; e>=8 -> K = base + 16 + h*8 + (e-8)
#pragma unroll
        for (int e = 0; e < 8; ++e) {
            af[e]     = yrow[base + h * 8 + e];
            af[8 + e] = yrow[base + 16 + h * 8 + e];
        }
        // B 32x16 f16 layout: e<8 -> K = base + h*8 + e ; e>=8 -> K = base + 16 + h*8 + (e-8)
#pragma unroll
        for (int e = 0; e < 8; ++e) {
            bf[e]     = grow[base + h * 8 + e];
            bf[8 + e] = grow[base + 16 + h * 8 + e];
        }

        acc = __builtin_amdgcn_wmma_f32_16x16x32_f16(
            /*neg_a=*/false, af, /*neg_b=*/false, bf,
            /*c_mod=*/(short)0, acc, /*reuse_a=*/false, /*reuse_b=*/false);
    }

    // C/D layout: lane half h, VGPR v -> row M = v + h*8, col N = lane&15
#pragma unroll
    for (int v = 0; v < 8; ++v) {
        int M = v + h * 8;
        Dot[((size_t)b * NTP + tt * 16 + M) * NPP + (pp * 16 + m)] = acc[v];
    }
}

// ---------------------------------------------------------------------------
// Per-pair costs: match (for Hungarian), cat_cost, attr_cost, box_cost, iou_arr
// ---------------------------------------------------------------------------
__global__ void pair_cost_kernel(const float* __restrict__ bbox, const float* __restrict__ box_preds,
                                 const float* __restrict__ cat_preds, const int* __restrict__ cat_idx,
                                 const float* __restrict__ S0, const float* __restrict__ Dot,
                                 float* __restrict__ mcost, float* __restrict__ ccost,
                                 float* __restrict__ acost, float* __restrict__ bcost,
                                 float* __restrict__ icost) {
    int idx = blockIdx.x * blockDim.x + threadIdx.x;
    const int total = BB * NT * NP;
    if (idx >= total) return;
    int p = idx % NP; int rem = idx / NP;
    int t = rem % NT; int b = rem / NT;

    int ci = cat_idx[b * NT + t];
    float pc = cat_preds[((size_t)b * NP + p) * CC + ci];
    float cat_match = 1.0f - pc;
    float cat_cost = -logf(pc + 1e-5f);

    float attr_cost = S0[b * NPP + p] + Dot[((size_t)b * NTP + t) * NPP + p];

    const float* bt = bbox      + ((size_t)b * NT + t) * 4;
    const float* bp = box_preds + ((size_t)b * NP + p) * 4;
    float ay1 = bt[0], ax1 = bt[1], ay2 = bt[2], ax2 = bt[3];
    float by1 = bp[0], bx1 = bp[1], by2 = bp[2], bx2 = bp[3];
    float area_a = fmaxf(ay2 - ay1, 0.f) * fmaxf(ax2 - ax1, 0.f);
    float area_b = fmaxf(by2 - by1, 0.f) * fmaxf(bx2 - bx1, 0.f);
    float inter  = fmaxf(fminf(ay2, by2) - fmaxf(ay1, by1), 0.f) *
                   fmaxf(fminf(ax2, bx2) - fmaxf(ax1, bx1), 0.f);
    float uni = area_a + area_b - inter;
    float iou = (uni > 0.f) ? inter / uni : 0.f;
    float enc = fmaxf(fmaxf(ay2, by2) - fminf(ay1, by1), 0.f) *
                fmaxf(fmaxf(ax2, bx2) - fminf(ax1, bx1), 0.f);
    float giou = iou - ((enc > 0.f) ? (enc - uni) / enc : 0.f);
    float l1 = 0.25f * (fabsf(ay1 - by1) + fabsf(ax1 - bx1) + fabsf(ay2 - by2) + fabsf(ax2 - bx2));
    float box_cost = 2.0f * (1.0f - giou) + 5.0f * l1;

    size_t off = ((size_t)b * NT + t) * NP + p;
    mcost[off] = cat_match + box_cost;
    ccost[off] = cat_cost;
    acost[off] = attr_cost;
    bcost[off] = box_cost;
    icost[off] = 1.0f - iou;
}

// ---------------------------------------------------------------------------
// Hungarian (Jonker-Volgenant), one block per batch, column-parallel.
// assign[b][j] (j=1..NP) = matched row+1 (0 = unmatched), matching reference p[].
// ---------------------------------------------------------------------------
__global__ void __launch_bounds__(256) hungarian_kernel(const float* __restrict__ cost,
                                                        const int* __restrict__ num_objects,
                                                        int* __restrict__ assign) {
    const int b = blockIdx.x;
    const int tid = threadIdx.x;
    const int m = NP;
    const double DINF = __builtin_inf();

    __shared__ double v_[NP + 1];
    __shared__ double minv[NP + 1];
    __shared__ double u_[NT + 1];
    __shared__ int way[NP + 1];
    __shared__ int p_[NP + 1];
    __shared__ int used[NP + 1];
    __shared__ double red_v[256];
    __shared__ int    red_j[256];
    __shared__ int j0_sh, i0_sh;
    __shared__ double delta_sh;

    int n = num_objects[b];
    if (n > NT) n = NT;

    for (int j = tid; j <= m; j += blockDim.x) { v_[j] = 0.0; p_[j] = 0; way[j] = 0; }
    for (int i = tid; i <= NT; i += blockDim.x) u_[i] = 0.0;
    __syncthreads();

    for (int i = 1; i <= n; ++i) {
        if (tid == 0) { p_[0] = i; j0_sh = 0; }
        for (int j = tid; j <= m; j += blockDim.x) { minv[j] = DINF; used[j] = 0; }
        __syncthreads();

        while (true) {
            int j0 = j0_sh;
            if (tid == 0) { used[j0] = 1; i0_sh = p_[j0]; }
            __syncthreads();
            int i0 = i0_sh;
            const float* crow = cost + ((size_t)b * NT + (i0 - 1)) * NP;
            double ui0 = u_[i0];

            // minv / way update over unused columns
            for (int j = 1 + tid; j <= m; j += blockDim.x) {
                if (!used[j]) {
                    double cur = (double)crow[j - 1] - ui0 - v_[j];
                    if (cur < minv[j]) { minv[j] = cur; way[j] = j0; }
                }
            }
            __syncthreads();

            // argmin over unused columns (lowest index wins ties, like np.argmin)
            double bestv = DINF; int bestj = 0x7fffffff;
            for (int j = 1 + tid; j <= m; j += blockDim.x) {
                double cv = used[j] ? DINF : minv[j];
                if (cv < bestv) { bestv = cv; bestj = j; }
            }
            red_v[tid] = bestv; red_j[tid] = bestj;
            __syncthreads();
            for (int s = 128; s > 0; s >>= 1) {
                if (tid < s) {
                    double ov = red_v[tid + s]; int oj = red_j[tid + s];
                    if (ov < red_v[tid] || (ov == red_v[tid] && oj < red_j[tid])) {
                        red_v[tid] = ov; red_j[tid] = oj;
                    }
                }
                __syncthreads();
            }
            if (tid == 0) { delta_sh = red_v[0]; j0_sh = red_j[0]; }
            __syncthreads();
            double delta = delta_sh;

            // potentials update (used[] still reflects pre-j1 state, incl. j0)
            for (int j = tid; j <= m; j += blockDim.x) {
                if (used[j]) { u_[p_[j]] += delta; v_[j] -= delta; }
                else         { minv[j] -= delta; }
            }
            __syncthreads();
            if (p_[j0_sh] == 0) break;   // uniform: shared state, post-barrier
        }

        // augmenting path (serial)
        if (tid == 0) {
            int j0 = j0_sh;
            while (j0 != 0) { int j1 = way[j0]; p_[j0] = p_[j1]; j0 = j1; }
        }
        __syncthreads();
    }

    for (int j = tid; j <= m; j += blockDim.x) assign[b * (NP + 1) + j] = p_[j];
}

// ---------------------------------------------------------------------------
// Finalize: deterministic single-block reductions -> 8 outputs
// ---------------------------------------------------------------------------
__device__ double block_sum(double val, double* buf) {
    int tid = threadIdx.x;
    buf[tid] = val;
    __syncthreads();
    for (int s = 128; s > 0; s >>= 1) {
        if (tid < s) buf[tid] += buf[tid + s];
        __syncthreads();
    }
    double r = buf[0];
    __syncthreads();
    return r;
}

__global__ void __launch_bounds__(256) finalize_kernel(const float* __restrict__ ccost,
                                                       const float* __restrict__ acost,
                                                       const float* __restrict__ bcost,
                                                       const float* __restrict__ icost,
                                                       const int* __restrict__ assign,
                                                       const int* __restrict__ num_objects,
                                                       const float* __restrict__ cat_preds,
                                                       float* __restrict__ out) {
    const int tid = threadIdx.x;
    __shared__ double buf[256];
    __shared__ float syp[BB];
    __shared__ double acc[8];   // cat, attr, box, iou, exist, c50, call, npred

    // total objects
    double tl = 0.0;
    for (int i = tid; i < BB; i += blockDim.x) tl += (double)num_objects[i];
    double tobj = block_sum(tl, buf);

    // per-batch sum of yp = 1 - cat_preds[b,p,0]
    for (int b = 0; b < BB; ++b) {
        double s = 0.0;
        for (int p = tid; p < NP; p += blockDim.x)
            s += (double)(1.0f - cat_preds[((size_t)b * NP + p) * CC + 0]);
        double r = block_sum(s, buf);
        if (tid == 0) syp[b] = (float)r;
        __syncthreads();
    }

    // assigned-pair sums
    double dcat = 0.0, dattr = 0.0, dbox = 0.0, diou = 0.0, dexist = 0.0;
    double c50 = 0.0, call = 0.0;
    for (int idx = tid; idx < BB * NP; idx += blockDim.x) {
        int p = idx % NP; int b = idx / NP;
        int r = assign[b * (NP + 1) + (p + 1)];
        if (r != 0) {
            int t = r - 1;
            size_t off = ((size_t)b * NT + t) * NP + p;
            dcat  += (double)ccost[off];
            dattr += (double)acost[off];
            dbox  += (double)bcost[off];
            float ia = icost[off];          // masked_iou value (= 1 - iou)
            diou += (double)ia;
            for (int k = 0; k < 10; ++k) {
                float thr = (50.0f + 5.0f * k) * 0.01f;
                if (ia >= thr) { call += 1.0; if (k == 0) c50 += 1.0; }
            }
            float yp = 1.0f - cat_preds[((size_t)b * NP + p) * CC + 0];
            float pn = yp / syp[b];
            pn = fminf(fmaxf(pn, EPSF), 1.0f - EPSF);
            dexist += (double)(-logf(pn));
        }
    }
    // num_predicted over cat_preds[:,1,:]
    double np_ = 0.0;
    for (int idx = tid; idx < BB * CC; idx += blockDim.x) {
        int c = idx % CC; int b = idx / CC;
        if (cat_preds[((size_t)b * NP + 1) * CC + c] < 0.5f) np_ += 1.0;
    }

    double r;
    r = block_sum(dcat,  buf); if (tid == 0) acc[0] = r;
    r = block_sum(dattr, buf); if (tid == 0) acc[1] = r;
    r = block_sum(dbox,  buf); if (tid == 0) acc[2] = r;
    r = block_sum(diou,  buf); if (tid == 0) acc[3] = r;
    r = block_sum(dexist,buf); if (tid == 0) acc[4] = r;
    r = block_sum(c50,   buf); if (tid == 0) acc[5] = r;
    r = block_sum(call,  buf); if (tid == 0) acc[6] = r;
    r = block_sum(np_,   buf); if (tid == 0) acc[7] = r;
    __syncthreads();

    if (tid == 0) {
        float to = (float)tobj;
        float category_cost  = (float)(acc[0]) / to;
        float attribute_cost = (float)(acc[1]) / to;
        float box_cost       = (float)(acc[2]) / to;
        float exist_loss     = (float)(acc[4]) / (float)(BB * NP);
        float total_loss     = category_cost + attribute_cost + box_cost + exist_loss;
        float iou_metric     = (float)(acc[3]) / to;
        float npred          = (float)(acc[7]);
        float mAP50          = (float)(acc[5]) / npred;
        float map5095        = (float)(acc[6]) / (npred * 10.0f);
        out[0] = total_loss; out[1] = category_cost; out[2] = attribute_cost;
        out[3] = box_cost;   out[4] = exist_loss;
        out[5] = iou_metric; out[6] = mAP50; out[7] = map5095;
    }
}

// ---------------------------------------------------------------------------
// Launch
// ---------------------------------------------------------------------------
extern "C" void kernel_launch(void* const* d_in, const int* in_sizes, int n_in,
                              void* d_out, int out_size, void* d_ws, size_t ws_size,
                              hipStream_t stream) {
    (void)in_sizes; (void)n_in; (void)out_size; (void)ws_size;
    const float* category   = (const float*)d_in[0];
    const float* attribute  = (const float*)d_in[1];
    const float* bbox       = (const float*)d_in[2];
    const int*   num_obj    = (const int*)  d_in[3];
    const float* cat_preds  = (const float*)d_in[4];
    const float* attr_preds = (const float*)d_in[5];
    const float* box_preds  = (const float*)d_in[6];
    float* out = (float*)d_out;

    char* base = (char*)d_ws;
    size_t off = 0;
    auto carve = [&](size_t bytes) -> char* {
        char* p = base + off;
        off += (bytes + 255) & ~(size_t)255;
        return p;
    };
    _Float16* Yh   = (_Float16*)carve((size_t)BB * NTP * AA * sizeof(_Float16));
    _Float16* Gh   = (_Float16*)carve((size_t)BB * NPP * AA * sizeof(_Float16));
    float*    S0   = (float*)   carve((size_t)BB * NPP * sizeof(float));
    int*      cidx = (int*)     carve((size_t)BB * NT * sizeof(int));
    float*    Dot  = (float*)   carve((size_t)BB * NTP * NPP * sizeof(float));
    float*    mcost= (float*)   carve((size_t)BB * NT * NP * sizeof(float));
    float*    ccost= (float*)   carve((size_t)BB * NT * NP * sizeof(float));
    float*    acost= (float*)   carve((size_t)BB * NT * NP * sizeof(float));
    float*    bcost= (float*)   carve((size_t)BB * NT * NP * sizeof(float));
    float*    icost= (float*)   carve((size_t)BB * NT * NP * sizeof(float));
    int*      assign=(int*)     carve((size_t)BB * (NP + 1) * sizeof(int));

    {
        int total = BB * NTP * AA;
        prep_y_kernel<<<(total + 255) / 256, 256, 0, stream>>>(attribute, Yh);
    }
    {
        int total = BB * NPP;
        prep_g_kernel<<<(total + 255) / 256, 256, 0, stream>>>(attr_preds, Gh, S0);
    }
    {
        int total = BB * NT;
        prep_catidx_kernel<<<(total + 255) / 256, 256, 0, stream>>>(category, cidx);
    }
    {
        int tiles = BB * (NTP / 16) * (NPP / 16);
        attr_wmma_kernel<<<tiles, 32, 0, stream>>>(Yh, Gh, Dot);
    }
    {
        int total = BB * NT * NP;
        pair_cost_kernel<<<(total + 255) / 256, 256, 0, stream>>>(
            bbox, box_preds, cat_preds, cidx, S0, Dot, mcost, ccost, acost, bcost, icost);
    }
    hungarian_kernel<<<BB, 256, 0, stream>>>(mcost, num_obj, assign);
    finalize_kernel<<<1, 256, 0, stream>>>(ccost, acost, bcost, icost, assign, num_obj, cat_preds, out);
}
